// DeepSeekMoE_69681549410914
// MI455X (gfx1250) — matrix operand order repaired and checked by reading, not compiled, not run on hardware
//
#include <hip/hip_runtime.h>
#include <math.h>

// Problem constants (from reference): B=2, S=2048, D=2048, F=1024, E=16, SH=2, K=2
#define TKN  4096      // B*S tokens
#define DIM  2048
#define FF   1024
#define NE   16
#define NSH  2
#define TOPK 2

#define MTILE 64                 // tokens per block (4 x 16-row WMMA sub-tiles)
#define MSUB  4
#define MTMAX (TKN / MTILE)      // 64 worst-case M tiles per expert
#define NCHUNK_UP 2              // FF columns split across blocks
#define NCHUNK_DN 4              // DIM columns split across blocks

typedef __attribute__((ext_vector_type(16))) __bf16 v16bf;
typedef __attribute__((ext_vector_type(8)))  float  v8f;

union Frag16 { uint4 q[2]; v16bf v; };

__device__ __forceinline__ unsigned short f2bf(float f) {
    unsigned int u = __float_as_uint(f);
    u += 0x7FFFu + ((u >> 16) & 1u);        // round-to-nearest-even
    return (unsigned short)(u >> 16);
}

__device__ __forceinline__ v8f wmma_bf16(v16bf a, v16bf b, v8f c) {
    return __builtin_amdgcn_wmma_f32_16x16x32_bf16(
        false, a, false, b, (short)0, c, false, false);
}

__device__ __forceinline__ float gelu_erf(float g) {
    return 0.5f * g * (1.f + erff(g * 0.70710678118654752f));
}

// ---------------------------------------------------------------- utilities

__global__ void zero_meta_kernel(int* counts, int* cursor) {
    int t = threadIdx.x;
    if (t < NE) { counts[t] = 0; cursor[t] = 0; }
}

__global__ void cvt_kernel(const float* __restrict__ in,
                           unsigned short* __restrict__ out, long n) {
    long i = (long)blockIdx.x * blockDim.x + threadIdx.x;
    if (i < n) out[i] = f2bf(in[i]);
}

// out[m][c][r] = bf16(in[m][r][c]); LDS-tiled 32x32 transpose, coalesced both ways.
__global__ __launch_bounds__(256)
void cvt_tr_kernel(const float* __restrict__ in,
                   unsigned short* __restrict__ out,
                   int R, int C, int tilesR, int tilesC) {
    int bt = blockIdx.x;
    const int tc = bt % tilesC; bt /= tilesC;
    const int tr = bt % tilesR;
    const int m  = bt / tilesR;
    __shared__ float t[32][33];
    const int lx = threadIdx.x & 31;
    const int ly = threadIdx.x >> 5;          // 0..7
    const long base = (long)m * R * C;
    const float* ip = in + base + (long)(tr * 32) * C + tc * 32;
    #pragma unroll
    for (int i = 0; i < 32; i += 8)
        t[ly + i][lx] = ip[(long)(ly + i) * C + lx];
    __syncthreads();
    unsigned short* op = out + base + (long)(tc * 32) * R + tr * 32;
    #pragma unroll
    for (int i = 0; i < 32; i += 8)
        op[(long)(ly + i) * R + lx] = f2bf(t[lx][ly + i]);
}

__global__ void init_out_kernel(const float* __restrict__ x,
                                float* __restrict__ out, long n) {
    long i = (long)blockIdx.x * blockDim.x + threadIdx.x;
    if (i < n) out[i] = x[i];
}

// ---------------------------------------------------------------- router

__global__ __launch_bounds__(32)
void router_kernel(const float* __restrict__ x, const float* __restrict__ wa,
                   int* __restrict__ eidx, float* __restrict__ ew,
                   int* __restrict__ counts) {
    int t    = blockIdx.x;
    int lane = threadIdx.x;
    const float* xr = x + (size_t)t * DIM;

    float acc[NE];
    #pragma unroll
    for (int e = 0; e < NE; ++e) acc[e] = 0.f;

    for (int d = lane; d < DIM; d += 32) {
        float xv = xr[d];
        const float* wr = wa + (size_t)d * NE;
        #pragma unroll
        for (int e = 0; e < NE; ++e) acc[e] += xv * wr[e];
    }
    #pragma unroll
    for (int e = 0; e < NE; ++e) {
        float v = acc[e];
        #pragma unroll
        for (int o = 16; o > 0; o >>= 1) v += __shfl_xor(v, o, 32);
        acc[e] = v;
    }
    if (lane == 0) {
        float aff[NE];
        #pragma unroll
        for (int e = 0; e < NE; ++e) aff[e] = 1.f / (1.f + expf(-acc[e]));
        int i0 = 0;
        for (int e = 1; e < NE; ++e) if (aff[e] > aff[i0]) i0 = e;
        int i1 = -1;
        for (int e = 0; e < NE; ++e)
            if (e != i0 && (i1 < 0 || aff[e] > aff[i1])) i1 = e;
        float s = aff[i0] + aff[i1];
        eidx[t * 2 + 0] = i0;  ew[t * 2 + 0] = aff[i0] / s;
        eidx[t * 2 + 1] = i1;  ew[t * 2 + 1] = aff[i1] / s;
        atomicAdd(&counts[i0], 1);
        atomicAdd(&counts[i1], 1);
    }
}

__global__ void scan_kernel(const int* __restrict__ counts,
                            int* __restrict__ offsets) {
    if (threadIdx.x == 0) {
        int run = 0;
        for (int e = 0; e < NE; ++e) { offsets[e] = run; run += counts[e]; }
    }
}

__global__ void scatter_kernel(const int* __restrict__ eidx,
                               const float* __restrict__ ew,
                               const int* __restrict__ offsets,
                               int* __restrict__ cursor,
                               int* __restrict__ tok_list,
                               float* __restrict__ w_list) {
    int t = blockIdx.x * blockDim.x + threadIdx.x;
    if (t >= TKN) return;
    #pragma unroll
    for (int k = 0; k < TOPK; ++k) {
        int e   = eidx[t * 2 + k];
        int pos = atomicAdd(&cursor[e], 1);
        tok_list[offsets[e] + pos] = t;
        w_list[offsets[e] + pos]   = ew[t * 2 + k];
    }
}

// ---------------------------------------------------------------- up GEMM
// y[slot][FF] = gelu(x*G + gb) * (x*W1 + b1), bf16, gathered 64-token M tiles.
// Per wave: 1 N-tile, 4 M sub-tiles sharing each B fragment (8 WMMA / k-step).
__global__ __launch_bounds__(256)
void moe_up_kernel(const unsigned short* __restrict__ xb,
                   const unsigned short* __restrict__ gt,   // [E][FF][DIM]
                   const unsigned short* __restrict__ w1t,  // [E][FF][DIM]
                   const float* __restrict__ gb,
                   const float* __restrict__ b1,
                   const int*   __restrict__ tok_list,
                   const int*   __restrict__ counts,
                   const int*   __restrict__ offsets,
                   unsigned short* __restrict__ y,
                   int shared_mode) {
    int bx = blockIdx.x;
    const int chunk = bx % NCHUNK_UP; bx /= NCHUNK_UP;
    const int mtile = bx % MTMAX;
    const int e     = bx / MTMAX;
    const int cnt   = shared_mode ? TKN : counts[e];
    const int row0  = mtile * MTILE;
    if (row0 >= cnt) return;
    int valid = cnt - row0; if (valid > MTILE) valid = MTILE;
    const long slot0 = shared_mode ? ((long)TKN * TOPK + (long)e * TKN + row0)
                                   : ((long)offsets[e] + row0);

    __shared__ int tok_s[MTILE];
    const int tid = threadIdx.x;
    if (tid < MTILE) {
        if (shared_mode)      tok_s[tid] = row0 + tid;
        else if (tid < valid) tok_s[tid] = tok_list[offsets[e] + row0 + tid];
        else                  tok_s[tid] = tok_list[offsets[e] + row0];  // pad
    }
    __syncthreads();

    const int wave = tid >> 5, lane = tid & 31;
    const int laneHi = lane >> 4, ln = lane & 15;
    const int koffA = laneHi * 16, koffB = laneHi * 16;

    const unsigned short* ap[MSUB];
    #pragma unroll
    for (int s = 0; s < MSUB; ++s)
        ap[s] = xb + (size_t)tok_s[s * 16 + ln] * DIM;

    const unsigned short* gte = gt  + (size_t)e * FF * DIM;
    const unsigned short* w1e = w1t + (size_t)e * FF * DIM;
    const float* gbe = gb + (size_t)e * FF;
    const float* b1e = b1 + (size_t)e * FF;
    const v8f vzero = {};

    const int ntPerChunk = FF / 16 / NCHUNK_UP;       // 32
    for (int j = 0; j < ntPerChunk / 8; ++j) {        // 4 tiles per wave
        const int nt   = chunk * ntPerChunk + j * 8 + wave;
        const int ncol = nt * 16 + ln;
        const unsigned short* bgp = gte + (size_t)ncol * DIM;
        const unsigned short* bwp = w1e + (size_t)ncol * DIM;
        v8f cg[MSUB], cw[MSUB];
        #pragma unroll
        for (int s = 0; s < MSUB; ++s) { cg[s] = vzero; cw[s] = vzero; }

        for (int kt = 0; kt < DIM / 32; ++kt) {
            const int ka = kt * 32 + koffA;
            const int kb = kt * 32 + koffB;
            Frag16 bg, bw;
            bg.q[0] = *(const uint4*)(bgp + kb);
            bg.q[1] = *(const uint4*)(bgp + kb + 8);
            bw.q[0] = *(const uint4*)(bwp + kb);
            bw.q[1] = *(const uint4*)(bwp + kb + 8);
            #pragma unroll
            for (int s = 0; s < MSUB; ++s) {
                Frag16 a;
                a.q[0] = *(const uint4*)(ap[s] + ka);
                a.q[1] = *(const uint4*)(ap[s] + ka + 8);
                cg[s] = wmma_bf16(a.v, bg.v, cg[s]);
                cw[s] = wmma_bf16(a.v, bw.v, cw[s]);
            }
            __builtin_prefetch(bgp + kb + 1024, 0, 1);   // global_prefetch ahead
            __builtin_prefetch(bwp + kb + 1024, 0, 1);
        }
        const float bgv = gbe[ncol];
        const float bwv = b1e[ncol];
        #pragma unroll
        for (int s = 0; s < MSUB; ++s) {
            #pragma unroll
            for (int i = 0; i < 8; ++i) {
                const int m = s * 16 + i + laneHi * 8;
                if (m < valid) {
                    float g  = cg[s][i] + bgv;
                    float h  = cw[s][i] + bwv;
                    y[(slot0 + m) * FF + ncol] = f2bf(gelu_erf(g) * h);
                }
            }
        }
    }
}

// ---------------------------------------------------------------- down GEMM
// out[tok] += weight * (y*W2 + b2). Per wave: 2 N-tiles x 4 M sub-tiles.
__global__ __launch_bounds__(256)
void moe_down_kernel(const unsigned short* __restrict__ y,
                     const unsigned short* __restrict__ w2t,  // [E][DIM][FF]
                     const float* __restrict__ b2,
                     const int*   __restrict__ tok_list,
                     const float* __restrict__ w_list,
                     const int*   __restrict__ counts,
                     const int*   __restrict__ offsets,
                     float* __restrict__ out,
                     int shared_mode) {
    int bx = blockIdx.x;
    const int chunk = bx % NCHUNK_DN; bx /= NCHUNK_DN;
    const int mtile = bx % MTMAX;
    const int e     = bx / MTMAX;
    const int cnt   = shared_mode ? TKN : counts[e];
    const int row0  = mtile * MTILE;
    if (row0 >= cnt) return;
    int valid = cnt - row0; if (valid > MTILE) valid = MTILE;
    const long slot0 = shared_mode ? ((long)TKN * TOPK + (long)e * TKN + row0)
                                   : ((long)offsets[e] + row0);

    __shared__ int   tok_s[MTILE];
    __shared__ float wgt_s[MTILE];
    const int tid = threadIdx.x;
    if (tid < MTILE) {
        if (shared_mode)      { tok_s[tid] = row0 + tid; wgt_s[tid] = 1.f; }
        else if (tid < valid) {
            int b = offsets[e] + row0 + tid;
            tok_s[tid] = tok_list[b]; wgt_s[tid] = w_list[b];
        } else                { tok_s[tid] = 0; wgt_s[tid] = 0.f; }
    }
    __syncthreads();

    const int wave = tid >> 5, lane = tid & 31;
    const int laneHi = lane >> 4, ln = lane & 15;
    const int koffA = laneHi * 16, koffB = laneHi * 16;

    const unsigned short* ap[MSUB];
    #pragma unroll
    for (int s = 0; s < MSUB; ++s)
        ap[s] = y + (size_t)(slot0 + s * 16 + ln) * FF;

    const unsigned short* w2e = w2t + (size_t)e * DIM * FF;
    const float* b2e = b2 + (size_t)e * DIM;
    const v8f vzero = {};

    const int ntPerChunk = DIM / 16 / NCHUNK_DN;      // 32
    for (int j = 0; j < ntPerChunk / 16; ++j) {       // 2 pairs per wave
        const int nt0 = chunk * ntPerChunk + j * 16 + wave * 2;
        const int nc0 = nt0 * 16 + ln;
        const int nc1 = nc0 + 16;
        const unsigned short* bp0 = w2e + (size_t)nc0 * FF;
        const unsigned short* bp1 = w2e + (size_t)nc1 * FF;
        v8f c0[MSUB], c1[MSUB];
        #pragma unroll
        for (int s = 0; s < MSUB; ++s) { c0[s] = vzero; c1[s] = vzero; }

        for (int kt = 0; kt < FF / 32; ++kt) {
            const int ka = kt * 32 + koffA;
            const int kb = kt * 32 + koffB;
            Frag16 f0, f1;
            f0.q[0] = *(const uint4*)(bp0 + kb);
            f0.q[1] = *(const uint4*)(bp0 + kb + 8);
            f1.q[0] = *(const uint4*)(bp1 + kb);
            f1.q[1] = *(const uint4*)(bp1 + kb + 8);
            #pragma unroll
            for (int s = 0; s < MSUB; ++s) {
                Frag16 a;
                a.q[0] = *(const uint4*)(ap[s] + ka);
                a.q[1] = *(const uint4*)(ap[s] + ka + 8);
                c0[s] = wmma_bf16(a.v, f0.v, c0[s]);
                c1[s] = wmma_bf16(a.v, f1.v, c1[s]);
            }
            __builtin_prefetch(bp0 + kb + 1024, 0, 1);
            __builtin_prefetch(bp1 + kb + 1024, 0, 1);
        }
        const float bias0 = b2e[nc0];
        const float bias1 = b2e[nc1];
        #pragma unroll
        for (int s = 0; s < MSUB; ++s) {
            #pragma unroll
            for (int i = 0; i < 8; ++i) {
                const int m = s * 16 + i + laneHi * 8;
                if (m < valid) {
                    const float w = wgt_s[m];
                    float* orow = out + (size_t)tok_s[m] * DIM;
                    atomicAdd(orow + nc0, w * (c0[s][i] + bias0));
                    atomicAdd(orow + nc1, w * (c1[s][i] + bias1));
                }
            }
        }
    }
}

// ---------------------------------------------------------------- host

extern "C" void kernel_launch(void* const* d_in, const int* in_sizes, int n_in,
                              void* d_out, int out_size, void* d_ws, size_t ws_size,
                              hipStream_t stream) {
    const float* x   = (const float*)d_in[0];
    const float* wa  = (const float*)d_in[1];
    const float* rg  = (const float*)d_in[2];
    const float* rgb = (const float*)d_in[3];
    const float* rw1 = (const float*)d_in[4];
    const float* rb1 = (const float*)d_in[5];
    const float* rw2 = (const float*)d_in[6];
    const float* rb2 = (const float*)d_in[7];
    const float* sg  = (const float*)d_in[8];
    const float* sgb = (const float*)d_in[9];
    const float* sw1 = (const float*)d_in[10];
    const float* sb1 = (const float*)d_in[11];
    const float* sw2 = (const float*)d_in[12];
    const float* sb2 = (const float*)d_in[13];
    float* out = (float*)d_out;

    size_t off = 0;
    auto alloc = [&](size_t bytes) {
        void* p = (char*)d_ws + off;
        off += (bytes + 255) & ~(size_t)255;
        return p;
    };
    unsigned short* xb    = (unsigned short*)alloc((size_t)TKN * DIM * 2);
    unsigned short* g_t   = (unsigned short*)alloc((size_t)NE * FF * DIM * 2);
    unsigned short* w1_t  = (unsigned short*)alloc((size_t)NE * FF * DIM * 2);
    unsigned short* w2_t  = (unsigned short*)alloc((size_t)NE * DIM * FF * 2);
    unsigned short* sg_t  = (unsigned short*)alloc((size_t)NSH * FF * DIM * 2);
    unsigned short* sw1_t = (unsigned short*)alloc((size_t)NSH * FF * DIM * 2);
    unsigned short* sw2_t = (unsigned short*)alloc((size_t)NSH * DIM * FF * 2);
    unsigned short* yws   = (unsigned short*)alloc((size_t)(TKN * TOPK + NSH * TKN) * FF * 2);
    int*   eidx     = (int*)  alloc((size_t)TKN * TOPK * 4);
    float* ew       = (float*)alloc((size_t)TKN * TOPK * 4);
    int*   tok_list = (int*)  alloc((size_t)TKN * TOPK * 4);
    float* w_list   = (float*)alloc((size_t)TKN * TOPK * 4);
    int*   counts   = (int*)  alloc(NE * 4);
    int*   offsets  = (int*)  alloc(NE * 4);
    int*   cursor   = (int*)  alloc(NE * 4);
    (void)ws_size; (void)in_sizes; (void)n_in; (void)out_size;

    const long nXD = (long)TKN * DIM;

    zero_meta_kernel<<<1, 64, 0, stream>>>(counts, cursor);
    cvt_kernel<<<(unsigned)((nXD + 255) / 256), 256, 0, stream>>>(x, xb, nXD);

    // gate/up: [D][F] -> [F][D] ; down: [F][D] -> [D][F]   (tiled transpose)
    {
        const int tRd = DIM / 32, tCf = FF / 32;   // 64, 32
        unsigned bR = (unsigned)(NE * tRd * tCf);
        cvt_tr_kernel<<<bR, 256, 0, stream>>>(rg,  g_t,  DIM, FF, tRd, tCf);
        cvt_tr_kernel<<<bR, 256, 0, stream>>>(rw1, w1_t, DIM, FF, tRd, tCf);
        cvt_tr_kernel<<<bR, 256, 0, stream>>>(rw2, w2_t, FF, DIM, tCf, tRd);
        unsigned bS = (unsigned)(NSH * tRd * tCf);
        cvt_tr_kernel<<<bS, 256, 0, stream>>>(sg,  sg_t,  DIM, FF, tRd, tCf);
        cvt_tr_kernel<<<bS, 256, 0, stream>>>(sw1, sw1_t, DIM, FF, tRd, tCf);
        cvt_tr_kernel<<<bS, 256, 0, stream>>>(sw2, sw2_t, FF, DIM, tCf, tRd);
    }

    router_kernel<<<TKN, 32, 0, stream>>>(x, wa, eidx, ew, counts);
    scan_kernel<<<1, 32, 0, stream>>>(counts, offsets);
    scatter_kernel<<<TKN / 256, 256, 0, stream>>>(eidx, ew, offsets, cursor,
                                                  tok_list, w_list);

    init_out_kernel<<<(unsigned)((nXD + 255) / 256), 256, 0, stream>>>(x, out, nXD);

    // Up-projection (gated, gelu) into y workspace
    moe_up_kernel<<<NE * MTMAX * NCHUNK_UP, 256, 0, stream>>>(
        xb, g_t, w1_t, rgb, rb1, tok_list, counts, offsets, yws, 0);
    moe_up_kernel<<<NSH * MTMAX * NCHUNK_UP, 256, 0, stream>>>(
        xb, sg_t, sw1_t, sgb, sb1, nullptr, nullptr, nullptr, yws, 1);

    // Down-projection + weighted scatter-add into d_out (residual already there)
    moe_down_kernel<<<NE * MTMAX * NCHUNK_DN, 256, 0, stream>>>(
        yws, w2_t, rb2, tok_list, w_list, counts, offsets, out, 0);
    moe_down_kernel<<<NSH * MTMAX * NCHUNK_DN, 256, 0, stream>>>(
        yws, sw2_t, sb2, nullptr, nullptr, nullptr, nullptr, out, 1);
}
